// Argphy_70377334112430
// MI455X (gfx1250) — matrix operand order repaired and checked by reading, not compiled, not run on hardware
//
#include <hip/hip_runtime.h>

// ---------------- problem constants ----------------
#define B_TOT   1024
#define T_STEPS 64
#define H       64
#define WD      128
#define DIN     67
#define KIN     96      // DIN padded to a multiple of 32 for WMMA K-tiles
#define TILE    16      // batch rows per wave (WMMA M)
#define K_SUB   4

#define LOG2E   1.44269504088896340736f
#define LN2     0.69314718055994530942f

typedef __attribute__((ext_vector_type(16))) _Float16 v16h;
typedef __attribute__((ext_vector_type(8)))  _Float16 v8h;
typedef __attribute__((ext_vector_type(8)))  float    v8f;

// Raw-hardware-transcendental activations (v_exp_f32/v_log_f32 are base-2).
// softplus(x) = log1p(exp(x)) == max(x,0) + ln2*log2(1 + exp2(-log2e*|x|)).
// Argument of log2 is in [1,2] -> no denormal guard needed; exp2 arg <= 0 -> no overflow.
static __device__ __forceinline__ float softplusf(float x) {
    const float e = __builtin_amdgcn_exp2f(fabsf(x) * -LOG2E);
    const float l = __builtin_amdgcn_logf(1.f + e);
    return fmaf(l, LN2, fmaxf(x, 0.f));
}
static __device__ __forceinline__ float sigmoidf_(float x) {
    return __builtin_amdgcn_rcpf(1.f + __builtin_amdgcn_exp2f(x * -LOG2E));
}
static __device__ __forceinline__ float fast_tanhf(float x) {
    // tanh(x) = 1 - 2/(exp(2x)+1)
    const float e = __builtin_amdgcn_exp2f(x * (2.f * LOG2E));
    return fmaf(-2.f, __builtin_amdgcn_rcpf(e + 1.f), 1.f);
}

static __device__ __forceinline__ v16h cat16(v8h lo, v8h hi) {
    return __builtin_shufflevector(lo, hi, 0,1,2,3,4,5,6,7,8,9,10,11,12,13,14,15);
}

// A fragment: 16x32 f16 tile, rows = batch (M), from LDS row-major [16][stride]
// ISA layout: lanes 0-15 -> M=lane, K = kk+{0..7, 16..23}; lanes 16-31 -> +8
static __device__ __forceinline__ v16h frag_A(const _Float16* z, int stride, int kk, int lane) {
    const int m  = lane & 15;
    const int k0 = kk + ((lane & 16) ? 8 : 0);
    const _Float16* p = z + m * stride + k0;
    v8h lo = *(const v8h*)(p);
    v8h hi = *(const v8h*)(p + 16);
    return cat16(lo, hi);
}

// B fragment: 32x16 f16 tile (KxN). Weights stored as W[n][k] (k contiguous),
// which IS B(k,n)=W[n,k] for out = z @ W^T.
// ISA layout: lanes 0-15 -> N=lane, K = kk+{0..7, 16..23}; lanes 16-31 -> +8
static __device__ __forceinline__ v16h frag_B(const _Float16* w, int stride, int nn, int kk, int lane) {
    const int n  = nn + (lane & 15);
    const int k0 = kk + ((lane & 16) ? 8 : 0);
    const _Float16* p = w + n * stride + k0;
    v8h lo = *(const v8h*)(p);
    v8h hi = *(const v8h*)(p + 16);
    return cat16(lo, hi);
}

// GEMM layer: [16 x (32*KT)] @ W^T -> softplus -> f16 ping-pong buffer [16][WD]
template<int KT, int NT, int ZSTRIDE, int WSTRIDE>
static __device__ __forceinline__ void layer_sp(const _Float16* zin, const _Float16* W,
                                                const float* bias, _Float16* zout, int lane) {
#pragma unroll
    for (int nt = 0; nt < NT; ++nt) {
        const int nn = nt * 16;
        const int nl = nn + (lane & 15);
        const float bv = bias[nl];
        v8f c;
#pragma unroll
        for (int r = 0; r < 8; ++r) c[r] = bv;
#pragma unroll
        for (int kt = 0; kt < KT; ++kt) {
            v16h a = frag_A(zin, ZSTRIDE, kt * 32, lane);
            v16h b = frag_B(W, WSTRIDE, nn, kt * 32, lane);
            c = __builtin_amdgcn_wmma_f32_16x16x32_f16(false, a, false, b, (short)0, c, false, false);
        }
        const int mb = (lane & 16) ? 8 : 0;
#pragma unroll
        for (int r = 0; r < 8; ++r)
            zout[(mb + r) * WD + nl] = (_Float16)softplusf(c[r]);
    }
}

// Final layer: [16x128] @ W3^T -> dh = sdyn * tanh(1e-4 * x) -> f32 [16][H]
static __device__ __forceinline__ void layer_final(const _Float16* zin, const _Float16* W,
                                                   const float* bias, float sdyn,
                                                   float (*dh)[H], int lane) {
#pragma unroll
    for (int nt = 0; nt < 4; ++nt) {
        const int nn = nt * 16;
        const int nl = nn + (lane & 15);
        const float bv = bias[nl];
        v8f c;
#pragma unroll
        for (int r = 0; r < 8; ++r) c[r] = bv;
#pragma unroll
        for (int kt = 0; kt < 4; ++kt) {
            v16h a = frag_A(zin, WD, kt * 32, lane);
            v16h b = frag_B(W, WD, nn, kt * 32, lane);
            c = __builtin_amdgcn_wmma_f32_16x16x32_f16(false, a, false, b, (short)0, c, false, false);
        }
        const int mb = (lane & 16) ? 8 : 0;
#pragma unroll
        for (int r = 0; r < 8; ++r)
            dh[mb + r][nl] = sdyn * fast_tanhf(1.0e-4f * c[r]);
    }
}

// One rhs evaluation for this wave's 16 rows
static __device__ __forceinline__ void rhs_eval(
    int lane,
    const float (*nin)[4], const float (*hin)[H],
    float (*dns)[4], float (*dh)[H],
    _Float16 (*zI)[KIN], _Float16 (*zA)[WD], _Float16 (*zB)[WD],
    const _Float16* w0, const _Float16* w1, const _Float16* w2, const _Float16* w3,
    const float* sb0, const float* sb1, const float* sb2, const float* sb3,
    const float* sWeta, float beta, float sdyn,
    float ll, float rr, float Nn, float dd, float cc)
{
    // --- eta = sigmoid(h . Weta + beta): all 32 lanes, 2 half-sums per row ---
    {
        const int m  = lane & 15;
        const int jo = (lane & 16) ? 32 : 0;
        float acc = 0.f;
#pragma unroll 8
        for (int j = 0; j < 32; ++j) acc += hin[m][jo + j] * sWeta[jo + j];
        acc += __shfl_xor(acc, 16);
        if (lane < TILE) {
            const float ee = sigmoidf_(acc + beta);
            const float Tu = nin[m][0] * 1000.f;
            const float Ti = nin[m][1] * 100.f;
            const float V  = nin[m][2] * 100000.f;
            const float dTu = ll - rr * Tu - ee * Tu * V;
            const float dTi = ee * Tu * V - dd * Ti;
            const float dV  = Nn * dd * Ti - cc * V;
            dns[m][0] = dTu * (1.f / 1000.f);
            dns[m][1] = dTi * (1.f / 100.f);
            dns[m][2] = dV  * (1.f / 100000.f);
        }
    }
    // --- build MLP input [h(64), ns(3)] as f16 (pad 67..95 pre-zeroed once) ---
    for (int i = lane; i < TILE * H; i += 32) {
        const int m = i >> 6, j = i & 63;
        zI[m][j] = (_Float16)hin[m][j];
    }
    for (int i = lane; i < TILE * 3; i += 32) {
        const int m = i / 3, c2 = i % 3;
        zI[m][H + c2] = (_Float16)nin[m][c2];
    }
    __syncthreads();
    layer_sp<3, 8, KIN, KIN>(&zI[0][0], w0, sb0, &zB[0][0], lane);
    __syncthreads();
    layer_sp<4, 8, WD, WD>(&zB[0][0], w1, sb1, &zA[0][0], lane);
    __syncthreads();
    layer_sp<4, 8, WD, WD>(&zA[0][0], w2, sb2, &zB[0][0], lane);
    __syncthreads();
    layer_final(&zB[0][0], w3, sb3, sdyn, dh, lane);
    __syncthreads();
}

// accumulate k into RK sums and build next stage input
static __device__ __forceinline__ void acc_stage(
    int lane, float w, float stagec,
    float (*hacc)[H], float (*nacc)[4],
    const float (*dh)[H], const float (*dns)[4],
    const float (*h0)[H], const float (*n0)[4],
    float (*hin)[H], float (*nin)[4])
{
    for (int i = lane; i < TILE * H; i += 32) {
        const int m = i >> 6, j = i & 63;
        const float k = dh[m][j];
        hacc[m][j] += w * k;
        hin[m][j]  = h0[m][j] + stagec * k;
    }
    for (int i = lane; i < TILE * 3; i += 32) {
        const int m = i / 3, c2 = i % 3;
        const float k = dns[m][c2];
        nacc[m][c2] += w * k;
        nin[m][c2]  = n0[m][c2] + stagec * k;
    }
}

// ---------------- weight prep: f32 -> f16 (padded) into workspace ----------------
__global__ void prep_weights(const float* W0, const float* W1, const float* W2, const float* W3,
                             _Float16* ws) {
    _Float16* w0 = ws;                      // [128][96]
    _Float16* w1 = w0 + WD * KIN;           // [128][128]
    _Float16* w2 = w1 + WD * WD;            // [128][128]
    _Float16* w3 = w2 + WD * WD;            // [64][128]
    const int tid  = blockIdx.x * blockDim.x + threadIdx.x;
    const int nthr = gridDim.x * blockDim.x;
    for (int i = tid; i < WD * KIN; i += nthr) {
        const int n = i / KIN, k = i % KIN;
        w0[i] = (k < DIN) ? (_Float16)W0[n * DIN + k] : (_Float16)0.f;
    }
    for (int i = tid; i < WD * WD; i += nthr) {
        w1[i] = (_Float16)W1[i];
        w2[i] = (_Float16)W2[i];
    }
    for (int i = tid; i < H * WD; i += nthr) w3[i] = (_Float16)W3[i];
}

// ---------------- main kernel: one wave integrates 16 trajectories ----------------
__global__ __launch_bounds__(32) void ode_wmma_kernel(
    const float* __restrict__ y0, const float* __restrict__ ts,
    const float* __restrict__ scale_dyn_p,
    const float* __restrict__ b0, const float* __restrict__ b1,
    const float* __restrict__ b2, const float* __restrict__ b3,
    const float* __restrict__ hidden_vec, const float* __restrict__ Weta,
    const float* __restrict__ beta_p, const float* __restrict__ parameter,
    const _Float16* __restrict__ wsw, float* __restrict__ out)
{
    __shared__ alignas(16) _Float16 zI[TILE][KIN];
    __shared__ alignas(16) _Float16 zA[TILE][WD];
    __shared__ alignas(16) _Float16 zB[TILE][WD];
    __shared__ float sh[TILE][H], shin[TILE][H], sdh[TILE][H], shacc[TILE][H];
    __shared__ float sns[TILE][4], snin[TILE][4], sdns[TILE][4], snacc[TILE][4];
    __shared__ float sb0[WD], sb1[WD], sb2[WD], sb3[H], sWetaL[H];

    const int lane  = threadIdx.x;
    const int bbase = blockIdx.x * TILE;

    const _Float16* w0 = wsw;
    const _Float16* w1 = w0 + WD * KIN;
    const _Float16* w2 = w1 + WD * WD;
    const _Float16* w3 = w2 + WD * WD;

    // stage biases / eta weights in LDS; zero layer-0 input pad once
    for (int i = lane; i < WD; i += 32) { sb0[i] = b0[i]; sb1[i] = b1[i]; sb2[i] = b2[i]; }
    for (int i = lane; i < H;  i += 32) { sb3[i] = b3[i]; sWetaL[i] = Weta[i]; }
    for (int i = lane; i < TILE * KIN; i += 32) (&zI[0][0])[i] = (_Float16)0.f;

    const float beta = beta_p[0];
    const float sdyn = scale_dyn_p[0];
    const float ll = softplusf(parameter[0]);
    const float rr = softplusf(parameter[1]);
    const float Nn = softplusf(parameter[2]);
    const float dd = softplusf(parameter[3]);
    const float cc = softplusf(parameter[4]);

    const float rnsc[3] = {1.f / 1000.f, 1.f / 100.f, 1.f / 100000.f};

    // initial state
    for (int i = lane; i < TILE * H; i += 32) { const int m = i >> 6, j = i & 63; sh[m][j] = hidden_vec[j]; }
    for (int i = lane; i < TILE * 3; i += 32) {
        const int m = i / 3, c2 = i % 3;
        sns[m][c2] = y0[(bbase + m) * 3 + c2] * rnsc[c2];
    }
    __syncthreads();

    float* outh = out + (size_t)B_TOT * T_STEPS * 3;
    // t = 0 snapshot
    for (int i = lane; i < TILE * 3; i += 32) {
        const int m = i / 3, c2 = i % 3;
        out[(size_t)(bbase + m) * T_STEPS * 3 + c2] = sns[m][c2];
    }
    for (int i = lane; i < TILE * H; i += 32) {
        const int m = i >> 6, j = i & 63;
        outh[(size_t)(bbase + m) * T_STEPS * H + j] = sh[m][j];
    }

#pragma unroll 1
    for (int ti = 1; ti < T_STEPS; ++ti) {
        const float dt  = ts[ti] - ts[ti - 1];
        const float dts = dt * (1.f / K_SUB);
#pragma unroll 1
        for (int sub = 0; sub < K_SUB; ++sub) {
            // zero RK accumulators
            for (int i = lane; i < TILE * H; i += 32) shacc[i >> 6][i & 63] = 0.f;
            for (int i = lane; i < TILE * 4; i += 32) snacc[i >> 2][i & 3]  = 0.f;
            __syncthreads();

            // k1
            rhs_eval(lane, sns, sh, sdns, sdh, zI, zA, zB, w0, w1, w2, w3,
                     sb0, sb1, sb2, sb3, sWetaL, beta, sdyn, ll, rr, Nn, dd, cc);
            acc_stage(lane, 1.f, 0.5f * dts, shacc, snacc, sdh, sdns, sh, sns, shin, snin);
            __syncthreads();
            // k2
            rhs_eval(lane, snin, shin, sdns, sdh, zI, zA, zB, w0, w1, w2, w3,
                     sb0, sb1, sb2, sb3, sWetaL, beta, sdyn, ll, rr, Nn, dd, cc);
            acc_stage(lane, 2.f, 0.5f * dts, shacc, snacc, sdh, sdns, sh, sns, shin, snin);
            __syncthreads();
            // k3
            rhs_eval(lane, snin, shin, sdns, sdh, zI, zA, zB, w0, w1, w2, w3,
                     sb0, sb1, sb2, sb3, sWetaL, beta, sdyn, ll, rr, Nn, dd, cc);
            acc_stage(lane, 2.f, dts, shacc, snacc, sdh, sdns, sh, sns, shin, snin);
            __syncthreads();
            // k4
            rhs_eval(lane, snin, shin, sdns, sdh, zI, zA, zB, w0, w1, w2, w3,
                     sb0, sb1, sb2, sb3, sWetaL, beta, sdyn, ll, rr, Nn, dd, cc);
            // state update: y += dts/6 * (k1 + 2k2 + 2k3 + k4)
            const float s6 = dts * (1.f / 6.f);
            for (int i = lane; i < TILE * H; i += 32) {
                const int m = i >> 6, j = i & 63;
                sh[m][j] += s6 * (shacc[m][j] + sdh[m][j]);
            }
            for (int i = lane; i < TILE * 3; i += 32) {
                const int m = i / 3, c2 = i % 3;
                sns[m][c2] += s6 * (snacc[m][c2] + sdns[m][c2]);
            }
            __syncthreads();
        }
        // snapshot at t = ti
        for (int i = lane; i < TILE * 3; i += 32) {
            const int m = i / 3, c2 = i % 3;
            out[(size_t)(bbase + m) * T_STEPS * 3 + (size_t)ti * 3 + c2] = sns[m][c2];
        }
        for (int i = lane; i < TILE * H; i += 32) {
            const int m = i >> 6, j = i & 63;
            outh[(size_t)(bbase + m) * T_STEPS * H + (size_t)ti * H + j] = sh[m][j];
        }
    }
}

extern "C" void kernel_launch(void* const* d_in, const int* in_sizes, int n_in,
                              void* d_out, int out_size, void* d_ws, size_t ws_size,
                              hipStream_t stream) {
    const float* y0         = (const float*)d_in[0];
    const float* ts         = (const float*)d_in[1];
    const float* scale_dyn  = (const float*)d_in[2];
    const float* W0         = (const float*)d_in[3];
    const float* b0         = (const float*)d_in[4];
    const float* W1         = (const float*)d_in[5];
    const float* b1         = (const float*)d_in[6];
    const float* W2         = (const float*)d_in[7];
    const float* b2         = (const float*)d_in[8];
    const float* W3         = (const float*)d_in[9];
    const float* b3         = (const float*)d_in[10];
    const float* hidden_vec = (const float*)d_in[11];
    const float* Weta       = (const float*)d_in[12];
    const float* beta       = (const float*)d_in[13];
    const float* parameter  = (const float*)d_in[14];
    _Float16* wsw = (_Float16*)d_ws;

    prep_weights<<<64, 256, 0, stream>>>(W0, W1, W2, W3, wsw);
    ode_wmma_kernel<<<B_TOT / TILE, 32, 0, stream>>>(
        y0, ts, scale_dyn, b0, b1, b2, b3, hidden_vec, Weta, beta, parameter,
        wsw, (float*)d_out);
}
